// TransformerForMultipleChoice_Fusion_Layer_45741401702655
// MI455X (gfx1250) — matrix operand order repaired and checked by reading, not compiled, not run on hardware
//
#include <hip/hip_runtime.h>
#include <hip/hip_bf16.h>
#include <math.h>

// ---------------------------------------------------------------------------
// Types for CDNA5 WMMA (wave32): bf16 A/B fragments, f32 accumulator.
// ---------------------------------------------------------------------------
typedef __attribute__((ext_vector_type(16))) __bf16 v16bf;
typedef __attribute__((ext_vector_type(8)))  float  v8f;

// A-fragment K index (contiguous order, matches B fills):
// lanes 0-15: K 0..15 ; lanes 16-31: K 16..31
__device__ __forceinline__ int kIdxA(int e, int half) {
    return e + 16 * half;
}

// ---------------------------------------------------------------------------
// f32 -> bf16 conversion (grid-stride)
// ---------------------------------------------------------------------------
__global__ __launch_bounds__(256) void cvt_kernel(
    const float* __restrict__ s, __bf16* __restrict__ d, size_t n) {
    size_t i = (size_t)blockIdx.x * blockDim.x + threadIdx.x;
    size_t stride = (size_t)gridDim.x * blockDim.x;
    for (; i < n; i += stride) d[i] = (__bf16)s[i];
}

// f32 [K][N] -> bf16 [N][K] transposed conversion, per layer (blockIdx.y = l)
__global__ __launch_bounds__(256) void cvtT_kernel(
    const float* __restrict__ s, __bf16* __restrict__ d, int K, int N) {
    size_t KN = (size_t)K * N;
    const float* sl = s + (size_t)blockIdx.y * KN;
    __bf16* dl = d + (size_t)blockIdx.y * KN;
    size_t i = (size_t)blockIdx.x * blockDim.x + threadIdx.x;
    size_t stride = (size_t)gridDim.x * blockDim.x;
    for (; i < KN; i += stride) {
        int nn = (int)(i / K);
        int kk = (int)(i % K);
        dl[i] = (__bf16)sl[(size_t)kk * N + nn];   // write-coalesced transpose
    }
}

// ---------------------------------------------------------------------------
// Fusion-weight kernel: w = softmax((mean_S x) @ lin_w + lin_b) per layer.
// ---------------------------------------------------------------------------
__global__ __launch_bounds__(256) void fusew_kernel(
    const float* __restrict__ x, const float* __restrict__ lw,
    const float* __restrict__ lb, float* __restrict__ wf) {
    int n = blockIdx.x;
    int tid = threadIdx.x;
    __shared__ float xm[768];
    __shared__ float w12[12];
    for (int h = tid; h < 768; h += 256) {
        float s = 0.f;
        for (int ss = 0; ss < 256; ++ss)
            s += x[((size_t)n * 256 + ss) * 768 + h];
        xm[h] = s * (1.f / 256.f);
    }
    __syncthreads();
    if (tid < 12) {
        float acc = lb[tid];
        for (int h = 0; h < 768; ++h) acc += xm[h] * lw[h * 12 + tid];
        w12[tid] = acc;
    }
    __syncthreads();
    if (tid < 3) {
        int l = tid;
        float m = w12[4 * l];
        for (int j = 1; j < 4; ++j) m = fmaxf(m, w12[4 * l + j]);
        float e[4], sum = 0.f;
        for (int j = 0; j < 4; ++j) { e[j] = __expf(w12[4 * l + j] - m); sum += e[j]; }
        for (int j = 0; j < 4; ++j) wf[((size_t)l * 16 + n) * 4 + j] = e[j] / sum;
    }
}

// ---------------------------------------------------------------------------
// Bias blend: bias[l,n,q,k] = sum_m w[l,n,m] * mask_m[n,q,k]
// ---------------------------------------------------------------------------
__global__ __launch_bounds__(256) void bias_kernel(
    const float* __restrict__ wf, const float* __restrict__ cm,
    const float* __restrict__ dep, const float* __restrict__ ent,
    const float* __restrict__ sent, float* __restrict__ bias) {
    int idx = blockIdx.x;          // l*4096 + n*256 + q
    int l = idx >> 12;
    int n = (idx >> 8) & 15;
    int qq = idx & 255;
    int k = threadIdx.x;
    const float* w = wf + ((size_t)l * 16 + n) * 4;
    size_t base = ((size_t)n * 256 + qq) * 256 + k;
    float v = w[0] * cm[base] + w[1] * dep[base]
            + w[2] * ent[(size_t)n * 256 + k] + w[3] * sent[base];
    bias[(size_t)idx * 256 + k] = v;
}

// ---------------------------------------------------------------------------
// WMMA bf16 GEMM: C = A[M,K](bf16) @ Bt[N,K](bf16, pre-transposed) + biasv
// Dual epilogue: optional f32 C and/or bf16 Cbf.  Optional ReLU.
// Block 256 thr = 8 waves (2 M x 4 N); 128x128 tile, K-step 32.
// LDS pitch 40 bf16 = 80 B = 20 dwords: fragment segments 16B-aligned
// (-> ds_load_b128) and 20*r mod 64 distinct for r=0..15 (conflict-free).
// ---------------------------------------------------------------------------
#define BM 128
#define BN 128
#define BK 32
#define LP (BK + 8)   // LDS pitch in bf16 (16B-aligned segments)

__global__ __launch_bounds__(256) void gemm_bf16(
    const __bf16* __restrict__ A, const __bf16* __restrict__ Bt,
    const float* __restrict__ biasv, float* __restrict__ C,
    __bf16* __restrict__ Cbf, int M, int N, int K, int relu) {
    __shared__ __bf16 As[BM][LP];
    __shared__ __bf16 Bs[BN][LP];   // Bs[col][k] : K-contiguous per column

    int tid = threadIdx.x;
    int lane = tid & 31, w = tid >> 5;
    int r = lane & 15, half = lane >> 4;
    int bm = blockIdx.y * BM, bn = blockIdx.x * BN;
    int waveM = (w & 1) * 64, waveN = (w >> 1) * 32;

    v8f acc[4][2];
#pragma unroll
    for (int mt = 0; mt < 4; ++mt)
#pragma unroll
        for (int nt = 0; nt < 2; ++nt)
#pragma unroll
            for (int i = 0; i < 8; ++i) acc[mt][nt][i] = 0.f;

    // staging coordinates (both sides 16 contiguous bf16 = 32B per thread)
    int arow = tid >> 1, ac0 = (tid & 1) * 16;
    int bcol = tid & 127, bk0 = (tid >> 7) * 16;

    for (int k0 = 0; k0 < K; k0 += BK) {
        {
            const __bf16* src = A + (size_t)(bm + arow) * K + k0 + ac0;
#pragma unroll
            for (int j = 0; j < 16; ++j) As[arow][ac0 + j] = src[j];
        }
        {
            const __bf16* src = Bt + (size_t)(bn + bcol) * K + k0 + bk0;
#pragma unroll
            for (int j = 0; j < 16; ++j) Bs[bcol][bk0 + j] = src[j];
        }
        if (k0 + BK < K) {   // prefetch next K-tile (global_prefetch_b8)
            __builtin_prefetch(A + (size_t)(bm + arow) * K + k0 + BK + ac0, 0, 3);
            __builtin_prefetch(Bt + (size_t)(bn + bcol) * K + k0 + BK + bk0, 0, 3);
        }
        __syncthreads();

        v16bf af[4], bfr[2];
#pragma unroll
        for (int mt = 0; mt < 4; ++mt) {
            int row = waveM + mt * 16 + r;
#pragma unroll
            for (int e = 0; e < 16; ++e) af[mt][e] = As[row][kIdxA(e, half)];
        }
#pragma unroll
        for (int nt = 0; nt < 2; ++nt) {
            int col = waveN + nt * 16 + r;
#pragma unroll
            for (int e = 0; e < 16; ++e) bfr[nt][e] = Bs[col][e + 16 * half];
        }
#pragma unroll
        for (int mt = 0; mt < 4; ++mt)
#pragma unroll
            for (int nt = 0; nt < 2; ++nt)
                acc[mt][nt] = __builtin_amdgcn_wmma_f32_16x16x32_bf16(
                    false, af[mt], false, bfr[nt], (short)0, acc[mt][nt],
                    false, false);
        __syncthreads();
    }

    // epilogue: C layout per ISA (VGPR i -> M = i + 8*half, N = lane&15)
#pragma unroll
    for (int mt = 0; mt < 4; ++mt)
#pragma unroll
        for (int nt = 0; nt < 2; ++nt)
#pragma unroll
            for (int i = 0; i < 8; ++i) {
                int row = bm + waveM + mt * 16 + i + 8 * half;
                int col = bn + waveN + nt * 16 + r;
                float v = acc[mt][nt][i] + biasv[col];
                if (relu) v = fmaxf(v, 0.f);
                if (C)   C[(size_t)row * N + col] = v;
                if (Cbf) Cbf[(size_t)row * N + col] = (__bf16)v;
            }
}

// ---------------------------------------------------------------------------
// Attention: block = (q-tile of 16 rows, head h, sequence n), 4 waves.
// Q/K/V arrive bf16. K (then V, LDS reuse) staged to LDS coalesced;
// QK^T and PV via WMMA; softmax in LDS; probs re-quantized bf16.
// Pitches: KVP=72 bf16 (36 dwords, distinct banks over 16 lanes, 16B-aligned
// segments); PP=264 bf16 (132 = 4 mod 64 dwords, distinct banks, aligned).
// ---------------------------------------------------------------------------
#define KVP 72
#define PP 264

__global__ __launch_bounds__(128) void attn_kernel(
    const __bf16* __restrict__ Qm, const __bf16* __restrict__ Km,
    const __bf16* __restrict__ Vm, const float* __restrict__ bias_l,
    __bf16* __restrict__ Om) {
    __shared__ __bf16 KV[256][KVP];   // K tile, then reused for V tile
    __shared__ float  Sld[16][256];
    __shared__ __bf16 Pld[16][PP];
    __shared__ float  red[16][8];
    __shared__ float  rowstat[16];

    int qt = blockIdx.x, h = blockIdx.y, n = blockIdx.z;
    int tid = threadIdx.x;
    int w = tid >> 5, lane = tid & 31, r = lane & 15, half = lane >> 4;
    const float scale = 0.125f;   // 1/sqrt(64)
    size_t qbase = (size_t)n * 256 + qt * 16;
    size_t hoff = (size_t)h * 64;

    // ---- stage K tile [256 keys][64 dims], coalesced bf16 copies ----
    for (int key = tid; key < 256; key += 128) {
        const __bf16* src = Km + ((size_t)n * 256 + key) * 768 + hoff;
#pragma unroll
        for (int d = 0; d < 64; ++d) KV[key][d] = src[d];
    }
    __syncthreads();

    // ---- scores: wave w handles ct = w, w+4, w+8, w+12 ----
    for (int ct = w; ct < 16; ct += 4) {
        v8f acc;
#pragma unroll
        for (int i = 0; i < 8; ++i) acc[i] = 0.f;
#pragma unroll
        for (int kk = 0; kk < 64; kk += 32) {
            v16bf af, bfr;
#pragma unroll
            for (int e = 0; e < 16; ++e) {
                af[e]  = Qm[(qbase + r) * 768 + hoff + kk + kIdxA(e, half)];
                bfr[e] = KV[ct * 16 + r][kk + e + 16 * half];
            }
            acc = __builtin_amdgcn_wmma_f32_16x16x32_bf16(
                false, af, false, bfr, (short)0, acc, false, false);
        }
#pragma unroll
        for (int i = 0; i < 8; ++i) {
            int mrow = i + 8 * half, col = ct * 16 + r;
            Sld[mrow][col] = acc[i] * scale + bias_l[(qbase + mrow) * 256 + col];
        }
    }
    __syncthreads();

    // ---- softmax: 8 threads per row, 32 cols each ----
    int row = tid & 15, part = tid >> 4;
    {
        float lmax = -1e30f;
        for (int c = part * 32; c < part * 32 + 32; ++c)
            lmax = fmaxf(lmax, Sld[row][c]);
        red[row][part] = lmax;
    }
    __syncthreads();
    if (part == 0) {
        float m = red[row][0];
        for (int j = 1; j < 8; ++j) m = fmaxf(m, red[row][j]);
        rowstat[row] = m;
    }
    __syncthreads();
    {
        float rmax = rowstat[row], lsum = 0.f;
        for (int c = part * 32; c < part * 32 + 32; ++c) {
            float e = __expf(Sld[row][c] - rmax);
            Sld[row][c] = e;
            lsum += e;
        }
        red[row][part] = lsum;
    }
    __syncthreads();
    if (part == 0) {
        float s = 0.f;
        for (int j = 0; j < 8; ++j) s += red[row][j];
        rowstat[row] = s;
    }
    __syncthreads();
    {
        float inv = 1.f / rowstat[row];
        for (int c = part * 32; c < part * 32 + 32; ++c)
            Pld[row][c] = (__bf16)(Sld[row][c] * inv);
    }
    __syncthreads();

    // ---- stage V tile into the same LDS (scores already consumed) ----
    for (int key = tid; key < 256; key += 128) {
        const __bf16* src = Vm + ((size_t)n * 256 + key) * 768 + hoff;
#pragma unroll
        for (int d = 0; d < 64; ++d) KV[key][d] = src[d];
    }
    __syncthreads();

    // ---- O = P(16x256) @ V(256x64): wave w owns N-tile nt = w ----
    {
        int nt = w;
        v8f acc;
#pragma unroll
        for (int i = 0; i < 8; ++i) acc[i] = 0.f;
        for (int kk = 0; kk < 256; kk += 32) {
            v16bf af, bfr;
#pragma unroll
            for (int e = 0; e < 16; ++e) {
                af[e]  = Pld[r][kk + kIdxA(e, half)];
                bfr[e] = KV[kk + e + 16 * half][nt * 16 + r];
            }
            acc = __builtin_amdgcn_wmma_f32_16x16x32_bf16(
                false, af, false, bfr, (short)0, acc, false, false);
        }
#pragma unroll
        for (int i = 0; i < 8; ++i)
            Om[(qbase + i + 8 * half) * 768 + hoff + nt * 16 + r] = (__bf16)acc[i];
    }
}

// ---------------------------------------------------------------------------
// Residual add + LayerNorm + pad-mask; dual f32 + bf16 output.
// One wave per 768-wide token row.
// ---------------------------------------------------------------------------
__global__ __launch_bounds__(256) void ln_kernel(
    const float* __restrict__ o, const float* __restrict__ resid,
    const float* __restrict__ g, const float* __restrict__ b,
    const int* __restrict__ ids, float* __restrict__ out,
    __bf16* __restrict__ outbf) {
    int token = blockIdx.x * 8 + (threadIdx.x >> 5);
    int lane = threadIdx.x & 31;
    const float* op = o + (size_t)token * 768;
    const float* rp = resid + (size_t)token * 768;
    float s = 0.f, q = 0.f;
    for (int j = lane; j < 768; j += 32) {
        float v = op[j] + rp[j];
        s += v; q += v * v;
    }
    for (int off = 16; off > 0; off >>= 1) {
        s += __shfl_xor(s, off, 32);
        q += __shfl_xor(q, off, 32);
    }
    float mean = s * (1.f / 768.f);
    float var = q * (1.f / 768.f) - mean * mean;
    float inv = rsqrtf(var + 1e-5f);
    float np = (ids[token] != 1) ? 1.f : 0.f;
    float* outp = out + (size_t)token * 768;
    __bf16* outbp = outbf + (size_t)token * 768;
    for (int j = lane; j < 768; j += 32) {
        float v = op[j] + rp[j];
        float y = (g[j] * (v - mean) * inv + b[j]) * np;
        outp[j] = y;
        outbp[j] = (__bf16)y;
    }
}

// ---------------------------------------------------------------------------
// Token BatchNorm stats over 4096 rows/feature; emit only token-0 seq rows.
// ---------------------------------------------------------------------------
__global__ __launch_bounds__(256) void bnt_kernel(
    const float* __restrict__ enc, const float* __restrict__ x,
    const float* __restrict__ g, const float* __restrict__ b,
    float* __restrict__ seq0) {
    int f = blockIdx.x, tid = threadIdx.x;
    __shared__ float s1[256], s2[256];
    float a = 0.f, q = 0.f;
    for (int row = tid; row < 4096; row += 256) {
        float v = enc[(size_t)row * 768 + f];
        a += v; q += v * v;
    }
    s1[tid] = a; s2[tid] = q;
    __syncthreads();
    for (int off = 128; off > 0; off >>= 1) {
        if (tid < off) { s1[tid] += s1[tid + off]; s2[tid] += s2[tid + off]; }
        __syncthreads();
    }
    float mean = s1[0] * (1.f / 4096.f);
    float var = s2[0] * (1.f / 4096.f) - mean * mean;
    float inv = rsqrtf(var + 1e-5f);
    if (tid < 16) {
        int n = tid;
        float e = enc[(size_t)(n * 256) * 768 + f];
        seq0[(size_t)n * 768 + f] =
            x[(size_t)(n * 256) * 768 + f] + g[f] * (e - mean) * inv + b[f];
    }
}

__global__ __launch_bounds__(128) void pool_kernel(
    const float* __restrict__ seq0, const float* __restrict__ pw,
    const float* __restrict__ pb, float* __restrict__ pooled) {
    int n = blockIdx.x;
    int j = blockIdx.y * 128 + threadIdx.x;
    float acc = pb[j];
    for (int h = 0; h < 768; ++h)
        acc += seq0[(size_t)n * 768 + h] * pw[(size_t)h * 768 + j];
    pooled[(size_t)n * 768 + j] = tanhf(acc);
}

__global__ __launch_bounds__(128) void bnb_kernel(
    const float* __restrict__ pooled, const float* __restrict__ g,
    const float* __restrict__ b, float* __restrict__ pbn) {
    int f = blockIdx.x * 128 + threadIdx.x;
    if (f >= 768) return;
    float m = 0.f;
    for (int n = 0; n < 16; ++n) m += pooled[(size_t)n * 768 + f];
    m *= (1.f / 16.f);
    float v = 0.f;
    for (int n = 0; n < 16; ++n) {
        float d = pooled[(size_t)n * 768 + f] - m;
        v += d * d;
    }
    v *= (1.f / 16.f);
    float inv = rsqrtf(v + 1e-5f);
    for (int n = 0; n < 16; ++n)
        pbn[(size_t)n * 768 + f] = g[f] * (pooled[(size_t)n * 768 + f] - m) * inv + b[f];
}

__global__ __launch_bounds__(256) void logits_kernel(
    const float* __restrict__ pbn, const float* __restrict__ cw,
    const float* __restrict__ cb, float* __restrict__ out) {
    __shared__ float redm[256];
    int n = blockIdx.x, tid = threadIdx.x;
    float s = 0.f;
    for (int h = tid; h < 768; h += 256) s += pbn[(size_t)n * 768 + h] * cw[h];
    redm[tid] = s;
    __syncthreads();
    for (int off = 128; off > 0; off >>= 1) {
        if (tid < off) redm[tid] += redm[tid + off];
        __syncthreads();
    }
    if (tid == 0) out[n] = redm[0] + cb[0];
}

// ---------------------------------------------------------------------------
// Host-side orchestration
// ---------------------------------------------------------------------------
extern "C" void kernel_launch(void* const* d_in, const int* in_sizes, int n_in,
                              void* d_out, int out_size, void* d_ws, size_t ws_size,
                              hipStream_t stream) {
    (void)in_sizes; (void)n_in; (void)out_size; (void)ws_size;
    const int*   ids   = (const int*)d_in[0];
    const float* x     = (const float*)d_in[1];
    // d_in[2] (pooled_output) is unused by the reference
    const float* cm    = (const float*)d_in[3];
    const float* dep   = (const float*)d_in[4];
    const float* ent   = (const float*)d_in[5];
    const float* sent  = (const float*)d_in[6];
    const float* lin_w = (const float*)d_in[7];
    const float* lin_b = (const float*)d_in[8];
    const float* Wq    = (const float*)d_in[9];
    const float* bq    = (const float*)d_in[10];
    const float* Wk    = (const float*)d_in[11];
    const float* bk    = (const float*)d_in[12];
    const float* Wv    = (const float*)d_in[13];
    const float* bv    = (const float*)d_in[14];
    const float* Wo    = (const float*)d_in[15];
    const float* bo    = (const float*)d_in[16];
    const float* ln1_g = (const float*)d_in[17];
    const float* ln1_b = (const float*)d_in[18];
    const float* W1    = (const float*)d_in[19];
    const float* b1    = (const float*)d_in[20];
    const float* W2    = (const float*)d_in[21];
    const float* b2    = (const float*)d_in[22];
    const float* ln2_g = (const float*)d_in[23];
    const float* ln2_b = (const float*)d_in[24];
    const float* bnt_g = (const float*)d_in[25];
    const float* bnt_b = (const float*)d_in[26];
    const float* pool_w = (const float*)d_in[27];
    const float* pool_b = (const float*)d_in[28];
    const float* bn_g  = (const float*)d_in[29];
    const float* bn_b  = (const float*)d_in[30];
    const float* cls_w = (const float*)d_in[31];
    const float* cls_b = (const float*)d_in[32];
    float* out = (float*)d_out;

    const size_t NT = 4096;   // N*S tokens
    const size_t H  = 768;
    const size_t HH = H * H;          // 589824
    const size_t HI = H * 4096;       // 3145728

    char* wsb = (char*)d_ws;
    size_t off = 0;
    auto carveF = [&](size_t nElem) -> float* {
        float* p = (float*)(wsb + off);
        off += (nElem * sizeof(float) + 255) & ~(size_t)255;
        return p;
    };
    auto carveB = [&](size_t nElem) -> __bf16* {
        __bf16* p = (__bf16*)(wsb + off);
        off += (nElem * sizeof(__bf16) + 255) & ~(size_t)255;
        return p;
    };
    // f32 buffers
    float* wf     = carveF(3 * 16 * 4);
    float* bias   = carveF((size_t)48 * 256 * 256);
    float* enc    = carveF(NT * H);
    float* enc2   = carveF(NT * H);
    float* ob     = carveF(NT * H);
    float* f2     = carveF(NT * H);
    float* seq0   = carveF(16 * H);
    float* pooled = carveF(16 * H);
    float* pbn    = carveF(16 * H);
    // bf16 buffers
    __bf16* encbf  = carveB(NT * H);
    __bf16* enc2bf = carveB(NT * H);
    __bf16* qbf    = carveB(NT * H);
    __bf16* kbf    = carveB(NT * H);
    __bf16* vbf    = carveB(NT * H);
    __bf16* abbf   = carveB(NT * H);
    __bf16* f1bf   = carveB(NT * 4096);
    __bf16* wqbf   = carveB(3 * HH);   // transposed [N][K] per layer
    __bf16* wkbf   = carveB(3 * HH);
    __bf16* wvbf   = carveB(3 * HH);
    __bf16* wobf   = carveB(3 * HH);
    __bf16* w1bf   = carveB(3 * HI);
    __bf16* w2bf   = carveB(3 * HI);

    // one-time weight conversion + transpose to [N][K] bf16
    cvtT_kernel<<<dim3(2048, 3), 256, 0, stream>>>(Wq, wqbf, 768, 768);
    cvtT_kernel<<<dim3(2048, 3), 256, 0, stream>>>(Wk, wkbf, 768, 768);
    cvtT_kernel<<<dim3(2048, 3), 256, 0, stream>>>(Wv, wvbf, 768, 768);
    cvtT_kernel<<<dim3(2048, 3), 256, 0, stream>>>(Wo, wobf, 768, 768);
    cvtT_kernel<<<dim3(2048, 3), 256, 0, stream>>>(W1, w1bf, 768, 4096);
    cvtT_kernel<<<dim3(2048, 3), 256, 0, stream>>>(W2, w2bf, 4096, 768);

    fusew_kernel<<<16, 256, 0, stream>>>(x, lin_w, lin_b, wf);
    bias_kernel<<<12288, 256, 0, stream>>>(wf, cm, dep, ent, sent, bias);
    hipMemcpyAsync(enc, x, NT * H * sizeof(float), hipMemcpyDeviceToDevice, stream);
    cvt_kernel<<<4096, 256, 0, stream>>>(x, encbf, NT * H);

    dim3 g768(768 / BN, 4096 / BM);    // (6, 32)
    dim3 g4096(4096 / BN, 4096 / BM);  // (32, 32)

    for (int l = 0; l < 3; ++l) {
        gemm_bf16<<<g768, 256, 0, stream>>>(encbf, wqbf + (size_t)l * HH,
            bq + (size_t)l * H, nullptr, qbf, 4096, 768, 768, 0);
        gemm_bf16<<<g768, 256, 0, stream>>>(encbf, wkbf + (size_t)l * HH,
            bk + (size_t)l * H, nullptr, kbf, 4096, 768, 768, 0);
        gemm_bf16<<<g768, 256, 0, stream>>>(encbf, wvbf + (size_t)l * HH,
            bv + (size_t)l * H, nullptr, vbf, 4096, 768, 768, 0);
        attn_kernel<<<dim3(16, 12, 16), 128, 0, stream>>>(
            qbf, kbf, vbf, bias + (size_t)l * 16 * 256 * 256, abbf);
        gemm_bf16<<<g768, 256, 0, stream>>>(abbf, wobf + (size_t)l * HH,
            bo + (size_t)l * H, ob, nullptr, 4096, 768, 768, 0);
        ln_kernel<<<512, 256, 0, stream>>>(ob, enc, ln1_g + (size_t)l * H,
            ln1_b + (size_t)l * H, ids, enc2, enc2bf);
        gemm_bf16<<<g4096, 256, 0, stream>>>(enc2bf, w1bf + (size_t)l * HI,
            b1 + (size_t)l * 4096, nullptr, f1bf, 4096, 4096, 768, 1);
        gemm_bf16<<<g768, 256, 0, stream>>>(f1bf, w2bf + (size_t)l * HI,
            b2 + (size_t)l * H, f2, nullptr, 4096, 768, 4096, 0);
        ln_kernel<<<512, 256, 0, stream>>>(f2, enc2, ln2_g + (size_t)l * H,
            ln2_b + (size_t)l * H, ids, enc, encbf);
    }

    bnt_kernel<<<768, 256, 0, stream>>>(enc, x, bnt_g, bnt_b, seq0);
    pool_kernel<<<dim3(16, 6), 128, 0, stream>>>(seq0, pool_w, pool_b, pooled);
    bnb_kernel<<<6, 128, 0, stream>>>(pooled, bn_g, bn_b, pbn);
    logits_kernel<<<16, 256, 0, stream>>>(pbn, cls_w, cls_b, out);
}
